// GraphVAE_44504451121672
// MI455X (gfx1250) — compile-verified, matrix operand order checked by reading, not run on hardware
//
#include <hip/hip_runtime.h>
#include <hip/hip_bf16.h>

// ---------------------------------------------------------------------------
// GraphVAE forward for MI455X (gfx1250, wave32, WMMA).
// HBM-bound on enc_w1/dec_w2 (2 x 201 MB f32): stream f32 weights with b128
// loads, convert to bf16 while staging into LDS, feed v_wmma_f32_16x16x32_bf16
// via b128 LDS fragment loads (documented per-lane layouts).
// ---------------------------------------------------------------------------

typedef __attribute__((ext_vector_type(16))) __bf16 v16bf;
typedef __attribute__((ext_vector_type(8)))  __bf16 v8bf;
typedef __attribute__((ext_vector_type(4)))  __bf16 v4bf;
typedef __attribute__((ext_vector_type(8)))  float  v8f;
typedef __attribute__((ext_vector_type(4)))  float  v4f;

#define BM 64
#define BN 64
#define BK 64
#define LDP 80   // LDS row pitch (elements): 160B, multiple of 16B, spreads banks

// ------------------------- JAX Threefry-2x32 RNG ---------------------------

__device__ __forceinline__ void threefry2x32(unsigned k0, unsigned k1,
                                             unsigned x0, unsigned x1,
                                             unsigned& o0, unsigned& o1) {
  const unsigned ks2 = k0 ^ k1 ^ 0x1BD11BDAu;
  x0 += k0; x1 += k1;
#define TF_R(r) { x0 += x1; x1 = (x1 << (r)) | (x1 >> (32 - (r))); x1 ^= x0; }
  TF_R(13) TF_R(15) TF_R(26) TF_R(6)
  x0 += k1;  x1 += ks2 + 1u;
  TF_R(17) TF_R(29) TF_R(16) TF_R(24)
  x0 += ks2; x1 += k0 + 2u;
  TF_R(13) TF_R(15) TF_R(26) TF_R(6)
  x0 += k0;  x1 += k1 + 3u;
  TF_R(17) TF_R(29) TF_R(16) TF_R(24)
  x0 += k1;  x1 += ks2 + 4u;
  TF_R(13) TF_R(15) TF_R(26) TF_R(6)
  x0 += ks2; x1 += k0 + 5u;
#undef TF_R
  o0 = x0; o1 = x1;
}

__device__ __forceinline__ void jax_fold(unsigned k0, unsigned k1, unsigned data,
                                         unsigned& o0, unsigned& o1) {
  threefry2x32(k0, k1, 0u, data, o0, o1);
}

// XLA single-precision ErfInv polynomial (what jax.random.normal lowers to).
__device__ __forceinline__ float erfinv_f32(float x) {
  float w = -__logf((1.0f - x) * (1.0f + x));
  float p;
  if (w < 5.0f) {
    w -= 2.5f;
    p = 2.81022636e-08f;
    p = 3.43273939e-07f  + p * w;
    p = -3.5233877e-06f  + p * w;
    p = -4.39150654e-06f + p * w;
    p = 0.00021858087f   + p * w;
    p = -0.00125372503f  + p * w;
    p = -0.00417768164f  + p * w;
    p = 0.246640727f     + p * w;
    p = 1.50140941f      + p * w;
  } else {
    w = sqrtf(w) - 3.0f;
    p = -0.000200214257f;
    p = 0.000100950558f + p * w;
    p = 0.00134934322f  + p * w;
    p = -0.00367342844f + p * w;
    p = 0.00573950773f  + p * w;
    p = -0.0076224613f  + p * w;
    p = 0.00943887047f  + p * w;
    p = 1.00167406f     + p * w;
    p = 2.83297682f     + p * w;
  }
  return p * x;
}

__device__ __forceinline__ float bits_to_normal(unsigned bits) {
  float f = __uint_as_float((bits >> 9) | 0x3f800000u) - 1.0f; // [0,1)
  const float lo = -0.99999994f;                               // nextafter(-1,0)
  float u = f * (1.0f - lo) + lo;
  u = fmaxf(u, lo);
  return 1.41421356237f * erfinv_f32(u);
}

__device__ __forceinline__ float jax_normal_at(unsigned k0, unsigned k1,
                                               unsigned idx, unsigned n) {
  unsigned half = n >> 1;
  unsigned pos = (idx < half) ? idx : (idx - half);
  unsigned o0, o1;
  threefry2x32(k0, k1, pos, half + pos, o0, o1);
  return bits_to_normal((idx < half) ? o0 : o1);
}

// --------------------------- GEMM building blocks --------------------------

// Stage A (BM x BK) f32 -> bf16 row-major in LDS.  Each thread: one row,
// 16 contiguous K-elements: 4x global b128 load, 2x ds_store_b128.
__device__ __forceinline__ void stage_A(const float* __restrict__ A, int lda,
                                        int m0, int kk, int tid,
                                        __bf16 (*As)[LDP]) {
  const int row = tid >> 2;
  const int cb  = (tid & 3) * 16;
  const float* p = &A[(long)(m0 + row) * lda + kk + cb];
  v4f a0 = *(const v4f*)(p + 0);
  v4f a1 = *(const v4f*)(p + 4);
  v4f a2 = *(const v4f*)(p + 8);
  v4f a3 = *(const v4f*)(p + 12);
  v8bf lo, hi;
#pragma unroll
  for (int i = 0; i < 4; ++i) {
    lo[i]     = (__bf16)a0[i];
    lo[4 + i] = (__bf16)a1[i];
    hi[i]     = (__bf16)a2[i];
    hi[4 + i] = (__bf16)a3[i];
  }
  *(v8bf*)&As[row][cb]     = lo;
  *(v8bf*)&As[row][cb + 8] = hi;
}

// Stage B (BK x BN) f32 -> bf16 TRANSPOSED in LDS (Bt[col][k]) so fragment
// reads are contiguous b128.  Each thread: 4 rows x 4 cols: 4x global b128
// load (coalesced along N), 4x ds_store_b64 of packed columns.
__device__ __forceinline__ void stage_B(const float* __restrict__ W, int ldw,
                                        int n0, int kk, int tid,
                                        __bf16 (*Bt)[LDP]) {
  const int r = (tid >> 4) * 4;
  const int c = (tid & 15) * 4;
  const float* p = &W[(long)(kk + r) * ldw + n0 + c];
  v4f w0 = *(const v4f*)(p);
  v4f w1 = *(const v4f*)(p + ldw);
  v4f w2 = *(const v4f*)(p + 2 * (long)ldw);
  v4f w3 = *(const v4f*)(p + 3 * (long)ldw);
#pragma unroll
  for (int i = 0; i < 4; ++i) {
    v4bf q;
    q[0] = (__bf16)w0[i];
    q[1] = (__bf16)w1[i];
    q[2] = (__bf16)w2[i];
    q[3] = (__bf16)w3[i];
    *(v4bf*)&Bt[c + i][r] = q;
  }
}

// Load one 16x32 operand fragment (A row-block or B^T col-block) for this
// lane per the documented 16-bit WMMA layout: two contiguous 8-element runs.
__device__ __forceinline__ v16bf frag16(const __bf16 (*T)[LDP], int rc,
                                        int kb, int half) {
  v8bf lo = *(const v8bf*)&T[rc][kb + 8 * half];
  v8bf hi = *(const v8bf*)&T[rc][kb + 16 + 8 * half];
  return __builtin_shufflevector(lo, hi, 0, 1, 2, 3, 4, 5, 6, 7,
                                 8, 9, 10, 11, 12, 13, 14, 15);
}

// ------------------------------ WMMA GEMM ----------------------------------
// C[M,N] = act(A[M,K] @ W[K,N] + bias[N]).  Block tile 64x64, 8 waves; each
// wave computes a 16x32 slab (A fragment reused across two WMMAs).  BK=64
// (two K-chunks per stage).  blockIdx.z = batch with element strides.
// act: 0 = none, 1 = relu, 2 = exp(0.5*x)

__global__ __launch_bounds__(256)
void gemm_wmma(const float* __restrict__ A, const float* __restrict__ W,
               const float* __restrict__ bias, float* __restrict__ C,
               int M, int N, int K,
               long sA, long sW, long sB, long sC, int act) {
  (void)M;
  __shared__ __align__(16) __bf16 As[BM][LDP];
  __shared__ __align__(16) __bf16 Bt[BN][LDP];
  const long z = blockIdx.z;
  A += z * sA; W += z * sW; C += z * sC;
  const float* bz = bias ? (bias + z * sB) : nullptr;

  const int m0 = blockIdx.y * BM;
  const int n0 = blockIdx.x * BN;
  const int tid  = threadIdx.x;
  const int lane = tid & 31;
  const int w    = tid >> 5;
  const int wm   = (w >> 1) << 4;   // 0/16/32/48
  const int wn   = (w & 1) << 5;    // 0/32 (two 16-col tiles each)
  const int half = lane >> 4;
  const int lm   = lane & 15;

  v8f acc0 = {};
  v8f acc1 = {};
  for (int kk = 0; kk < K; kk += BK) {
    stage_A(A, K, m0, kk, tid, As);
    stage_B(W, N, n0, kk, tid, Bt);
    if (kk + BK < K)  // global_prefetch_b8 on next weight tile
      __builtin_prefetch(&W[(long)(kk + BK) * N + n0 + (tid & 63)], 0, 1);
    __syncthreads();
#pragma unroll
    for (int kc = 0; kc < 2; ++kc) {
      const int kb = kc * 32;
      v16bf a  = frag16(As, wm + lm, kb, half);
      v16bf b0 = frag16(Bt, wn + lm, kb, half);
      v16bf b1 = frag16(Bt, wn + 16 + lm, kb, half);
      acc0 = __builtin_amdgcn_wmma_f32_16x16x32_bf16(
          false, a, false, b0, (short)0, acc0, false, false);
      acc1 = __builtin_amdgcn_wmma_f32_16x16x32_bf16(
          false, a, false, b1, (short)0, acc1, false, false);
    }
    __syncthreads();
  }

  // C layout: VGPR r -> M = wm + half*8 + r, N = wn(+16) + lm
#pragma unroll
  for (int r = 0; r < 8; ++r) {
    int row  = m0 + wm + half * 8 + r;
    int col0 = n0 + wn + lm;
    int col1 = col0 + 16;
    float v0 = acc0[r], v1 = acc1[r];
    if (bz) { v0 += bz[col0]; v1 += bz[col1]; }
    if (act == 1) { v0 = fmaxf(v0, 0.f); v1 = fmaxf(v1, 0.f); }
    else if (act == 2) { v0 = expf(0.5f * v0); v1 = expf(0.5f * v1); }
    C[(long)row * N + col0] = v0;
    C[(long)row * N + col1] = v1;
  }
}

// Split-K variant for the tall-skinny encoder L1 (K=49152): blockIdx.z is the
// K-slice; partial tiles accumulated with f32 atomics into pre-zeroed C.
__global__ __launch_bounds__(256)
void gemm_wmma_splitk(const float* __restrict__ A, const float* __restrict__ W,
                      float* __restrict__ C, int M, int N, int K, int kSlices) {
  (void)M;
  __shared__ __align__(16) __bf16 As[BM][LDP];
  __shared__ __align__(16) __bf16 Bt[BN][LDP];
  const int kPer = K / kSlices;
  const int kBeg = blockIdx.z * kPer;
  const int kEnd = kBeg + kPer;

  const int m0 = blockIdx.y * BM;
  const int n0 = blockIdx.x * BN;
  const int tid  = threadIdx.x;
  const int lane = tid & 31;
  const int w    = tid >> 5;
  const int wm   = (w >> 1) << 4;
  const int wn   = (w & 1) << 5;
  const int half = lane >> 4;
  const int lm   = lane & 15;

  v8f acc0 = {};
  v8f acc1 = {};
  for (int kk = kBeg; kk < kEnd; kk += BK) {
    stage_A(A, K, m0, kk, tid, As);
    stage_B(W, N, n0, kk, tid, Bt);
    if (kk + BK < kEnd)
      __builtin_prefetch(&W[(long)(kk + BK) * N + n0 + (tid & 63)], 0, 1);
    __syncthreads();
#pragma unroll
    for (int kc = 0; kc < 2; ++kc) {
      const int kb = kc * 32;
      v16bf a  = frag16(As, wm + lm, kb, half);
      v16bf b0 = frag16(Bt, wn + lm, kb, half);
      v16bf b1 = frag16(Bt, wn + 16 + lm, kb, half);
      acc0 = __builtin_amdgcn_wmma_f32_16x16x32_bf16(
          false, a, false, b0, (short)0, acc0, false, false);
      acc1 = __builtin_amdgcn_wmma_f32_16x16x32_bf16(
          false, a, false, b1, (short)0, acc1, false, false);
    }
    __syncthreads();
  }
#pragma unroll
  for (int r = 0; r < 8; ++r) {
    int row  = m0 + wm + half * 8 + r;
    int col0 = n0 + wn + lm;
    atomicAdd(&C[(long)row * N + col0], acc0[r]);
    atomicAdd(&C[(long)row * N + col0 + 16], acc1[r]);
  }
}

// ---------------------------- small kernels --------------------------------

__global__ void zero_f32(float* p, long n) {
  long i = blockIdx.x * (long)blockDim.x + threadIdx.x;
  if (i < n) p[i] = 0.f;
}

__global__ void bias_act(float* C, const float* __restrict__ bias,
                         long total, int N, int act) {
  long i = blockIdx.x * (long)blockDim.x + threadIdx.x;
  if (i >= total) return;
  float v = C[i] + bias[i % N];
  if (act == 1) v = fmaxf(v, 0.f);
  C[i] = v;
}

__global__ void init_prior(float* Pmu15, float* Pstd15) {
  int i = blockIdx.x * blockDim.x + threadIdx.x;
  if (i < 8192) { Pmu15[i] = 0.f; Pstd15[i] = 1.f; }
}

// Builds the padded (b, 960) TD input for step j, reproducing the reference's
// stack(K,b,D).reshape(b, K*D) reinterpretation + zero pad + Threefry eps.
__global__ void wp_pad(const float* __restrict__ Pmu, const float* __restrict__ Pstd,
                       const float* __restrict__ gs, float* __restrict__ P, int j) {
  long idx = blockIdx.x * (long)blockDim.x + threadIdx.x;
  const long total = 128L * 960;
  if (idx >= total) return;
  int r = (int)(idx / 960);
  int m = (int)(idx % 960);
  int Kn = 15 - j;                    // number of stacked nodes
  float val = 0.f;
  if (m < Kn * 64) {
    long s = (long)r * (Kn * 64) + m; // flat index into (K, b, D)
    int k  = (int)(s / 8192);         // 8192 = b*D
    int rem = (int)(s % 8192);
    int bb = rem / 64, d = rem % 64;
    int i  = j + 1 + k;
    unsigned f0, f1;
    jax_fold(0u, 42u, (unsigned)(j * 16 + i), f0, f1);
    float eps = jax_normal_at(f0, f1, (unsigned)(bb * 64 + d), 8192u);
    long o = (long)i * 8192 + bb * 64 + d;
    val = (Pmu[o] + Pstd[o] * eps) * gs[i * 16 + j];
  }
  P[idx] = val;
}

// Precision-weighted merge + the two 16-way products; writes z buffers.
__global__ void merge_z(const float* __restrict__ Pmu, const float* __restrict__ Pstd,
                        const float* __restrict__ mu_t, const float* __restrict__ std_t,
                        float* __restrict__ zpost_ws,
                        float* __restrict__ out_zpost, float* __restrict__ out_zprior) {
  int idx = blockIdx.x * blockDim.x + threadIdx.x;
  if (idx >= 8192) return;
  unsigned kpr0, kpr1, kpo0, kpo1;
  jax_fold(0u, 42u, 1000u, kpr0, kpr1);
  jax_fold(0u, 42u, 1001u, kpo0, kpo1);
  float zpr = 1.f, zpo = 1.f;
  for (int n = 0; n < 16; ++n) {
    long o = (long)n * 8192 + idx;
    float mh = Pmu[o],  sh = Pstd[o];
    float mt = mu_t[o], st = std_t[o];
    float ih = 1.f / (sh * sh), it = 1.f / (st * st);
    float var = 1.f / (ih + it);
    float mp  = var * (mh * ih + mt * it);
    float epr = jax_normal_at(kpr0, kpr1, (unsigned)o, 131072u);
    float epo = jax_normal_at(kpo0, kpo1, (unsigned)o, 131072u);
    zpr *= mh + sh * epr;
    zpo *= mp + var * epo;   // reference uses var_like directly as "std"
  }
  zpost_ws[idx] = zpo;
  out_zpost[idx] = zpo;
  out_zprior[idx] = zpr;
}

// ------------------------------- driver ------------------------------------

extern "C" void kernel_launch(void* const* d_in, const int* in_sizes, int n_in,
                              void* d_out, int out_size, void* d_ws, size_t ws_size,
                              hipStream_t stream) {
  (void)in_sizes; (void)n_in; (void)out_size; (void)ws_size;
  const float* x      = (const float*)d_in[0];
  const float* gs     = (const float*)d_in[1];
  const float* enc_w1 = (const float*)d_in[2];
  const float* enc_b1 = (const float*)d_in[3];
  const float* enc_w2 = (const float*)d_in[4];
  const float* enc_b2 = (const float*)d_in[5];
  const float* bu_w1  = (const float*)d_in[6];
  const float* bu_b1  = (const float*)d_in[7];
  const float* bu_wmu = (const float*)d_in[8];
  const float* bu_bmu = (const float*)d_in[9];
  const float* bu_wlv = (const float*)d_in[10];
  const float* bu_blv = (const float*)d_in[11];
  const float* td_w1  = (const float*)d_in[12];
  const float* td_b1  = (const float*)d_in[13];
  const float* td_wmu = (const float*)d_in[14];
  const float* td_bmu = (const float*)d_in[15];
  const float* td_wlv = (const float*)d_in[16];
  const float* td_blv = (const float*)d_in[17];
  const float* dec_w1 = (const float*)d_in[18];
  const float* dec_b1 = (const float*)d_in[19];
  const float* dec_w2 = (const float*)d_in[20];
  const float* dec_b2 = (const float*)d_in[21];
  float* out = (float*)d_out;

  // Workspace carve-up (floats)
  float* ws    = (float*)d_ws;
  float* enc_h = ws; ws += 131072;   // (128,1024)
  float* enc   = ws; ws += 65536;    // (128,512)
  float* hb    = ws; ws += 524288;   // (16,128,256)
  float* mu_t  = ws; ws += 131072;   // (16,128,64)
  float* std_t = ws; ws += 131072;
  float* Pmu   = ws; ws += 131072;   // prior mu  (16,128,64)
  float* Pstd  = ws; ws += 131072;   // prior std
  float* Ppad  = ws; ws += 122880;   // (128,960)
  float* h_td  = ws; ws += 32768;    // (128,256)
  float* zpost = ws; ws += 8192;     // (128,64)
  float* hd    = ws; ws += 131072;   // (128,1024)

  // ---- Encoder L1 (split-K over 16 slices -> 512 blocks) ----
  zero_f32<<<512, 256, 0, stream>>>(enc_h, 131072);
  gemm_wmma_splitk<<<dim3(1024 / BN, 128 / BM, 16), 256, 0, stream>>>(
      x, enc_w1, enc_h, 128, 1024, 49152, 16);
  bias_act<<<512, 256, 0, stream>>>(enc_h, enc_b1, 131072, 1024, 1);

  // ---- Encoder L2 ----
  gemm_wmma<<<dim3(512 / BN, 128 / BM, 1), 256, 0, stream>>>(
      enc_h, enc_w2, enc_b2, enc, 128, 512, 1024, 0, 0, 0, 0, 0);

  // ---- Bottom-up MLPs, batched over 16 nodes via grid.z ----
  gemm_wmma<<<dim3(256 / BN, 128 / BM, 16), 256, 0, stream>>>(
      enc, bu_w1, bu_b1, hb, 128, 256, 512,
      0L, 512L * 256, 256L, 128L * 256, 1);
  gemm_wmma<<<dim3(64 / BN, 128 / BM, 16), 256, 0, stream>>>(
      hb, bu_wmu, bu_bmu, mu_t, 128, 64, 256,
      128L * 256, 256L * 64, 64L, 8192L, 0);
  gemm_wmma<<<dim3(64 / BN, 128 / BM, 16), 256, 0, stream>>>(
      hb, bu_wlv, bu_blv, std_t, 128, 64, 256,
      128L * 256, 256L * 64, 64L, 8192L, 2);

  // ---- Top-down recursion (sequential, 15 steps) ----
  init_prior<<<32, 256, 0, stream>>>(Pmu + 15L * 8192, Pstd + 15L * 8192);
  for (int j = 14; j >= 0; --j) {
    wp_pad<<<(122880 + 255) / 256, 256, 0, stream>>>(Pmu, Pstd, gs, Ppad, j);
    gemm_wmma<<<dim3(256 / BN, 128 / BM, 1), 256, 0, stream>>>(
        Ppad, td_w1 + (long)j * 960 * 256, td_b1 + (long)j * 256, h_td,
        128, 256, 960, 0, 0, 0, 0, 1);
    gemm_wmma<<<dim3(64 / BN, 128 / BM, 1), 256, 0, stream>>>(
        h_td, td_wmu + (long)j * 256 * 64, td_bmu + (long)j * 64,
        Pmu + (long)j * 8192, 128, 64, 256, 0, 0, 0, 0, 0);
    gemm_wmma<<<dim3(64 / BN, 128 / BM, 1), 256, 0, stream>>>(
        h_td, td_wlv + (long)j * 256 * 64, td_blv + (long)j * 64,
        Pstd + (long)j * 8192, 128, 64, 256, 0, 0, 0, 0, 2);
  }

  // ---- Merge + latent products; z_posterior/z_prior to d_out tail ----
  merge_z<<<32, 256, 0, stream>>>(Pmu, Pstd, mu_t, std_t, zpost,
                                  out + 6291456L, out + 6291456L + 8192L);

  // ---- Decoder ----
  gemm_wmma<<<dim3(1024 / BN, 128 / BM, 1), 256, 0, stream>>>(
      zpost, dec_w1, dec_b1, hd, 128, 1024, 64, 0, 0, 0, 0, 1);
  gemm_wmma<<<dim3(49152 / BN, 128 / BM, 1), 256, 0, stream>>>(
      hd, dec_w2, dec_b2, out, 128, 49152, 1024, 0, 0, 0, 0, 0);
}
